// CausalSpatiotemporalModel_32521492365737
// MI455X (gfx1250) — matrix operand order repaired and checked by reading, not compiled, not run on hardware
//
#include <hip/hip_runtime.h>
#include <hip/hip_bf16.h>
#include <math.h>

// Problem constants (match reference module constants)
#define VN 10000
#define EN 50000
#define EPAD 50176      // EN rounded up past 128-row block overshoot
#define TT 4
#define SS 4
#define DD 128
#define PP 16

typedef __attribute__((ext_vector_type(16))) _Float16 v16h;
typedef __attribute__((ext_vector_type(8)))  _Float16 v8h;
typedef __attribute__((ext_vector_type(8)))  float    v8f;

// 16 contiguous floats via 4x b128 loads
__device__ __forceinline__ void vload16(const float* __restrict__ p, float* o) {
  const float4* q = (const float4*)p;
  float4 a = q[0], b = q[1], c = q[2], d = q[3];
  o[0]=a.x; o[1]=a.y; o[2]=a.z;  o[3]=a.w;
  o[4]=b.x; o[5]=b.y; o[6]=b.z;  o[7]=b.w;
  o[8]=c.x; o[9]=c.y; o[10]=c.z; o[11]=c.w;
  o[12]=d.x;o[13]=d.y;o[14]=d.z; o[15]=d.w;
}

// ---------------- A-matrix loaders (fused gathers) ----------------
// Contract: ld16(r,k) must be safe (finite values) for r < ceil(M/128)*128 and
// k < ceil(K/32)*32, and must return exact zeros for K <= k (ensured by padded
// buffers). k is always a multiple of 16.
struct LoadDense {           // stride is the (padded) row stride in floats
  const float* A; int stride;
  __device__ void ld16(int r, int k, float* o) const {
    vload16(A + (size_t)r * stride + k, o);
  }
};
struct LoadEdge {            // K=256: [h_frames[t,src] | h_frames[t,dst]], padded index arrays
  const float* H; const int* src; const int* dst;
  __device__ void ld16(int r, int k, float* o) const {
    int v = (k < 128) ? src[r] : dst[r];
    vload16(H + (size_t)v * 128 + (k & 127), o);
  }
};
struct LoadConcat {          // K=256: [A0 | A1], both r x 128
  const float* A0; const float* A1;
  __device__ void ld16(int r, int k, float* o) const {
    const float* p = (k < 128) ? (A0 + (size_t)r * 128 + k)
                               : (A1 + (size_t)r * 128 + (k - 128));
    vload16(p, o);
  }
};
struct LoadAttn {            // K=273: [h_t | all_msg[v,t,s] | tau | theta], r = v*SS+s
  const float* ht; const float* amsg; const float* theta; int t;
  __device__ void ld16(int r, int k, float* o) const {
    int v = r >> 2;
    int s = r & 3;
    if (k < 128) {
      vload16(ht + (size_t)v * 128 + k, o);
    } else if (k < 256) {
      vload16(amsg + (((size_t)v * TT + t) * SS + s) * 128 + (k - 128), o);
    } else {
#pragma unroll
      for (int i = 0; i < 16; ++i) {
        int kk = k + i;
        float val = 0.0f;
        if (kk == 256) val = (float)(t + 1);
        else if (kk < 273) val = theta[(size_t)v * PP + (kk - 257)];
        o[i] = val;
      }
    }
  }
};

// ---------------- C-matrix stores ----------------
struct StoreDense {
  float* C; int N;
  __device__ void st(int r, int n, float v) const { C[(size_t)r * N + n] = v; }
};
struct StoreMsg {            // r = t*VN + v  ->  all_msg[(v*TT+t)*SS+s, n]
  float* C; int s;
  __device__ void st(int r, int n, float v) const {
    int t = r / VN, vx = r - t * VN;
    C[(((size_t)vx * TT + t) * SS + s) * 128 + n] = v;
  }
};

// ---------------- generic WMMA GEMM:  C = act(A(MxK) * W(KxN) + bias) ----------------
// block = 256 threads (8 waves). Block tile 128(M) x 64(N). Wave tile 32x32 (2x2 wmma).
// N % 64 == 0. M ragged (store-guarded; loaders tolerate padded rows). K ragged
// (loaders provide zero padding; B fill uses clamp-and-select, no branches).
template <typename Loader, typename Store>
__global__ __launch_bounds__(256) void wmma_gemm_kernel(Loader loader,
                                                        const float* __restrict__ W,
                                                        const float* __restrict__ bias,
                                                        Store store,
                                                        int M, int K, int N, int act) {
  __shared__ __align__(16) _Float16 As[128][32];
  __shared__ __align__(16) _Float16 Bt[64][32];   // transposed: [n][k]

  const int tid  = threadIdx.x;
  const int lane = tid & 31;
  const int wid  = tid >> 5;   // 0..7
  const int wm   = wid & 3;    // 4 M-subwaves * 32 rows
  const int wn   = wid >> 2;   // 2 N-subwaves * 32 cols
  const int blockM = blockIdx.y * 128;
  const int blockN = blockIdx.x * 64;
  const int m16 = lane & 15;
  const int hi  = lane >> 4;

  // cooperative-fill assignments (each thread: one contiguous run)
  const int ar    = tid >> 1;          // A row 0..127
  const int akoff = (tid & 1) * 16;    // A k offset {0,16}
  const int bcol  = tid >> 2;          // B col 0..63
  const int bkoff = (tid & 3) * 8;     // B k offset {0,8,16,24}

  v8f acc[2][2];
#pragma unroll
  for (int ms = 0; ms < 2; ++ms)
#pragma unroll
    for (int ns = 0; ns < 2; ++ns)
#pragma unroll
      for (int i = 0; i < 8; ++i) acc[ms][ns][i] = 0.0f;

  for (int k0 = 0; k0 < K; k0 += 32) {
    // ---- A tile: 16 contiguous floats -> f16, two ds_store_b128 ----
    float t16[16];
    loader.ld16(blockM + ar, k0 + akoff, t16);
    v8h x0, x1;
#pragma unroll
    for (int j = 0; j < 8; ++j) { x0[j] = (_Float16)t16[j]; x1[j] = (_Float16)t16[8 + j]; }
    v8h* ap = (v8h*)&As[ar][akoff];
    ap[0] = x0; ap[1] = x1;

    // ---- B tile (transposed in LDS): clamp-and-select, branchless ----
    v8h bx;
#pragma unroll
    for (int i = 0; i < 8; ++i) {
      int gk = k0 + bkoff + i;
      int gkc = (gk < K) ? gk : 0;
      float v = W[(size_t)gkc * N + (blockN + bcol)];
      bx[i] = (_Float16)((gk < K) ? v : 0.0f);
    }
    *(v8h*)&Bt[bcol][bkoff] = bx;
    __syncthreads();

    // ---- fragments per ISA 7.12.2 layouts, vectorized from LDS ----
    v16h a[2], b[2];
#pragma unroll
    for (int ms = 0; ms < 2; ++ms) {
      const v8h* Ar = (const v8h*)&As[wm * 32 + ms * 16 + m16][0];
      v8h lo = Ar[hi];        // k = 8*hi   .. +7   (j = 0..7)
      v8h hv = Ar[2 + hi];    // k = 16+8*hi .. +7  (j = 8..15)
      a[ms] = __builtin_shufflevector(lo, hv, 0, 1, 2, 3, 4, 5, 6, 7,
                                      8, 9, 10, 11, 12, 13, 14, 15);
    }
#pragma unroll
    for (int ns = 0; ns < 2; ++ns) {
      const v8h* Br = (const v8h*)&Bt[wn * 32 + ns * 16 + m16][0];
      v8h lo = Br[hi];           // k = 8*hi      .. +7
      v8h hv = Br[2 + hi];       // k = 16+8*hi   .. +7
      b[ns] = __builtin_shufflevector(lo, hv, 0, 1, 2, 3, 4, 5, 6, 7,
                                      8, 9, 10, 11, 12, 13, 14, 15);
    }
#pragma unroll
    for (int ms = 0; ms < 2; ++ms)
#pragma unroll
      for (int ns = 0; ns < 2; ++ns)
        acc[ms][ns] = __builtin_amdgcn_wmma_f32_16x16x32_f16(
            false, a[ms], false, b[ns], (short)0, acc[ms][ns], false, false);
    __syncthreads();
  }

  // epilogue: bias + activation + store (C/D layout: lane n+16*hi, i -> M = i+8*hi)
#pragma unroll
  for (int ms = 0; ms < 2; ++ms)
#pragma unroll
    for (int ns = 0; ns < 2; ++ns) {
      int col = blockN + wn * 32 + ns * 16 + m16;
#pragma unroll
      for (int i = 0; i < 8; ++i) {
        int row = blockM + wm * 32 + ms * 16 + i + 8 * hi;
        if (row < M) {
          float v = acc[ms][ns][i] + bias[col];
          if (act == 1) v = fmaxf(v, 0.0f);
          else if (act == 2) v = tanhf(v);
          store.st(row, col, v);
        }
      }
    }
}

// ---------------- small scalar kernels (negligible FLOPs) ----------------
__global__ void pad_edges_kernel(const int* __restrict__ edges, int* __restrict__ eds) {
  int i = blockIdx.x * blockDim.x + threadIdx.x;
  if (i >= SS * 2 * EPAD) return;
  int seg = i / EPAD, e = i - seg * EPAD;
  eds[i] = (e < EN) ? edges[(size_t)seg * EN + e] : 0;
}

__global__ void build_inputs_kernel(const float* __restrict__ dynf,
                                    const float* __restrict__ reff,
                                    float* __restrict__ Xfeat, float* __restrict__ Xcur) {
  int i = blockIdx.x * blockDim.x + threadIdx.x;
  if (i < VN) {
#pragma unroll
    for (int c = 0; c < 32; ++c) {
      float v = 0.0f;
      if (c < 3)      v = dynf[(size_t)i * 3 + c];        // frame 0
      else if (c < 6) v = reff[(size_t)i * 3 + (c - 3)];
      Xcur[(size_t)i * 32 + c] = v;
    }
  }
  if (i < TT * VN) {
    int t = i / VN, v = i - t * VN;
#pragma unroll
    for (int c = 0; c < 32; ++c) {
      float x = 0.0f;
      if (c < 3)      x = dynf[(((size_t)(t + 1)) * VN + v) * 3 + c];
      else if (c < 6) x = reff[(((size_t)(t + 1)) * VN + v) * 3 + (c - 3)];
      Xfeat[(size_t)i * 32 + c] = x;
    }
  }
}

__global__ void prop_theta_kernel(const float* __restrict__ stiff, const float* __restrict__ mass,
                                  const int* __restrict__ constraint,
                                  const float* __restrict__ w0, const float* __restrict__ b0,
                                  const float* __restrict__ w1, const float* __restrict__ b1,
                                  float* __restrict__ theta) {
  int v = blockIdx.x * blockDim.x + threadIdx.x;
  if (v >= VN) return;
  float x0 = stiff[v], x1 = mass[v], x2 = (float)constraint[v];
  float h[64];
#pragma unroll
  for (int j = 0; j < 64; ++j)
    h[j] = fmaxf(x0 * w0[0 * 64 + j] + x1 * w0[1 * 64 + j] + x2 * w0[2 * 64 + j] + b0[j], 0.0f);
  for (int o = 0; o < PP; ++o) {
    float a = b1[o];
    for (int j = 0; j < 64; ++j) a += h[j] * w1[j * PP + o];
    theta[(size_t)v * PP + o] = a;
  }
}

__global__ void vel_kernel(const float* __restrict__ theta,
                           const float* __restrict__ w0, const float* __restrict__ b0,
                           const float* __restrict__ w1, const float* __restrict__ b1,
                           float* __restrict__ vout) {
  int v = blockIdx.x * blockDim.x + threadIdx.x;
  if (v >= VN) return;
  float a = b1[0];
  for (int j = 0; j < 64; ++j) {
    float hj = b0[j];
    for (int i = 0; i < PP; ++i) hj += theta[(size_t)v * PP + i] * w0[i * 64 + j];
    a += fmaxf(hj, 0.0f) * w1[j];
  }
  vout[v] = (a > 20.0f) ? a : log1pf(expf(a));   // softplus
}

__global__ void mask_kernel(const float* __restrict__ vvel, float* __restrict__ logmask) {
  int i = blockIdx.x * blockDim.x + threadIdx.x;
  if (i >= VN * TT) return;
  int v = i / TT, t = i - v * TT;
  float radius = vvel[v] * (float)(t + 1);
  float smax = 0.0f;
#pragma unroll
  for (int r = 1; r <= SS; ++r) smax += 1.0f / (1.0f + expf(-5.0f * (radius - (float)r)));
  smax = fmaxf(smax - 1.0f, 0.0f);
#pragma unroll
  for (int s = 0; s < SS; ++s) {
    float m = 1.0f / (1.0f + expf(-5.0f * (smax - (float)s)));
    logmask[((size_t)v * TT + t) * SS + s] = logf(m + 1e-9f);
  }
}

__global__ void scatter_add_kernel(const float* __restrict__ m, const int* __restrict__ dst,
                                   float* __restrict__ agg, int t) {
  int i = blockIdx.x * blockDim.x + threadIdx.x;
  if (i >= EN * DD) return;
  int e = i >> 7, d = i & 127;
  atomicAdd(&agg[(((size_t)t * VN) + dst[e]) * DD + d], m[(size_t)e * DD + d]);
}

__global__ void attn_logits_kernel(const float* __restrict__ hidden,
                                   const float* __restrict__ w1, const float* __restrict__ b1,
                                   const float* __restrict__ logmask, float* __restrict__ logits,
                                   int t) {
  int r = blockIdx.x * blockDim.x + threadIdx.x;   // r = v*SS + s
  if (r >= VN * SS) return;
  int v = r >> 2, s = r & 3;
  float a = b1[0];
  for (int d = 0; d < DD; ++d) a += hidden[(size_t)r * DD + d] * w1[d];
  size_t idx = ((size_t)v * TT + t) * SS + s;
  logits[idx] = a + logmask[idx];
}

__global__ void softmax_agg_kernel(const float* __restrict__ logits,
                                   const float* __restrict__ amsg,
                                   float* __restrict__ aggmsg) {
  __shared__ float w[TT * SS];
  int v = blockIdx.x;
  int tid = threadIdx.x;
  if (tid == 0) {
    float mx = -1e30f;
    for (int i = 0; i < TT * SS; ++i) mx = fmaxf(mx, logits[(size_t)v * TT * SS + i]);
    float ssum = 0.0f;
    for (int i = 0; i < TT * SS; ++i) {
      float e = expf(logits[(size_t)v * TT * SS + i] - mx);
      w[i] = e; ssum += e;
    }
    float inv = 1.0f / ssum;
    for (int i = 0; i < TT * SS; ++i) w[i] *= inv;
  }
  __syncthreads();
  int d = tid;   // 128 threads
  float a = 0.0f;
  for (int i = 0; i < TT * SS; ++i)
    a += w[i] * amsg[((size_t)v * TT * SS + i) * DD + d];
  aggmsg[(size_t)v * DD + d] = a;
}

__global__ void final_kernel(const float* __restrict__ h,
                             const float* __restrict__ w4, const float* __restrict__ b4,
                             const int* __restrict__ constraint, float* __restrict__ out) {
  int v = blockIdx.x * blockDim.x + threadIdx.x;
  if (v >= VN) return;
  float m = (constraint[v] == 0) ? 1.0f : 0.0f;
#pragma unroll
  for (int c = 0; c < 3; ++c) {
    float a = b4[c];
    for (int d = 0; d < DD; ++d) a += h[(size_t)v * DD + d] * w4[d * 3 + c];
    out[(size_t)v * 3 + c] = a * m;
  }
}

// ---------------- host launch ----------------
extern "C" void kernel_launch(void* const* d_in, const int* in_sizes, int n_in,
                              void* d_out, int out_size, void* d_ws, size_t ws_size,
                              hipStream_t stream) {
  (void)in_sizes; (void)n_in; (void)out_size; (void)ws_size;

  const int*   constraint = (const int*)d_in[0];
  const float* dynf       = (const float*)d_in[1];
  const float* reff       = (const float*)d_in[2];
  const float* stiff      = (const float*)d_in[3];
  const float* mass       = (const float*)d_in[4];
  const int*   edges      = (const int*)d_in[5];

  int p = 6;
  auto nx = [&]() { return (const float*)d_in[p++]; };
  const float *feat_w0 = nx(), *feat_b0 = nx(), *feat_w1 = nx(), *feat_b1 = nx();
  const float *prop_w0 = nx(), *prop_b0 = nx(), *prop_w1 = nx(), *prop_b1 = nx();
  const float *cur_w0  = nx(), *cur_b0  = nx(), *cur_w1  = nx(), *cur_b1  = nx();
  const float *vel_w0  = nx(), *vel_b0  = nx(), *vel_w1  = nx(), *vel_b1  = nx();
  const float *attn_w0 = nx(), *attn_b0 = nx(), *attn_w1 = nx(), *attn_b1 = nx();
  const float *dyn_w[5], *dyn_b[5];
  for (int i = 0; i < 5; ++i) { dyn_w[i] = nx(); dyn_b[i] = nx(); }
  const float *ew[SS][3], *eb[SS][3];
  for (int s = 0; s < SS; ++s)
    for (int l = 0; l < 3; ++l) { ew[s][l] = nx(); eb[s][l] = nx(); }
  const float *nw[SS][2], *nb[SS][2];
  for (int s = 0; s < SS; ++s)
    for (int l = 0; l < 2; ++l) { nw[s][l] = nx(); nb[s][l] = nx(); }

  // workspace layout (floats); every buffer gets 32K floats of tail slack so
  // padded-row reads (r up to ceil(M/128)*128-1) stay in-bounds. Peak ~193 MB.
  float* ws = (float*)d_ws;
  size_t off = 0;
  auto alloc = [&](size_t n) {
    float* q = ws + off;
    off += ((n + 32768) + 63) & ~(size_t)63;
    return q;
  };
  float* theta    = alloc((size_t)VN * PP);
  float* vvel     = alloc(VN);
  float* logmask  = alloc((size_t)VN * TT * SS);
  float* logits   = alloc((size_t)VN * TT * SS);
  float* Xfeat    = alloc((size_t)TT * VN * 32);      // stride 32, zero-padded cols
  float* Xcur     = alloc((size_t)VN * 32);
  float* h_frames = alloc((size_t)TT * VN * DD);
  float* h_t      = alloc((size_t)VN * DD);
  float* all_msg  = alloc((size_t)VN * TT * SS * DD);
  float* agg      = alloc((size_t)TT * VN * DD);
  float* agg_msg  = alloc((size_t)VN * DD);
  float* bufA     = alloc((size_t)EN * DD);
  float* bufB     = alloc((size_t)EN * DD);
  int*   eds      = (int*)alloc((size_t)SS * 2 * EPAD); // zero-padded edge index copy

  const dim3 blk(256);
  auto gg = [](int M, int N) { return dim3(N / 64, (M + 127) / 128); };

  // inputs + tiny heads
  pad_edges_kernel<<<(SS * 2 * EPAD + 255) / 256, 256, 0, stream>>>(edges, eds);
  build_inputs_kernel<<<(TT * VN + 255) / 256, 256, 0, stream>>>(dynf, reff, Xfeat, Xcur);
  prop_theta_kernel<<<(VN + 127) / 128, 128, 0, stream>>>(stiff, mass, constraint,
      prop_w0, prop_b0, prop_w1, prop_b1, theta);
  vel_kernel<<<(VN + 127) / 128, 128, 0, stream>>>(theta, vel_w0, vel_b0, vel_w1, vel_b1, vvel);
  mask_kernel<<<(VN * TT + 255) / 256, 256, 0, stream>>>(vvel, logmask);

  // h_frames = feat MLP (T*V rows)
  wmma_gemm_kernel<<<gg(TT * VN, DD), blk, 0, stream>>>(
      LoadDense{Xfeat, 32}, feat_w0, feat_b0, StoreDense{bufA, DD}, TT * VN, 6, DD, 1);
  wmma_gemm_kernel<<<gg(TT * VN, DD), blk, 0, stream>>>(
      LoadDense{bufA, DD}, feat_w1, feat_b1, StoreDense{h_frames, DD}, TT * VN, DD, DD, 0);

  // h_t = cur MLP
  wmma_gemm_kernel<<<gg(VN, DD), blk, 0, stream>>>(
      LoadDense{Xcur, 32}, cur_w0, cur_b0, StoreDense{bufA, DD}, VN, 6, DD, 1);
  wmma_gemm_kernel<<<gg(VN, DD), blk, 0, stream>>>(
      LoadDense{bufA, DD}, cur_w1, cur_b1, StoreDense{h_t, DD}, VN, DD, DD, 0);

  // multi-scale message passing (dominant compute; fused-gather WMMA GEMMs)
  for (int s = 0; s < SS; ++s) {
    const int* srcp = eds + (size_t)(s * 2 + 0) * EPAD;
    const int* dstp = eds + (size_t)(s * 2 + 1) * EPAD;
    const int* dst_raw = edges + (size_t)s * 2 * EN + EN;
    hipMemsetAsync(agg, 0, (size_t)TT * VN * DD * sizeof(float), stream);
    for (int t = 0; t < TT; ++t) {
      const float* Ht = h_frames + (size_t)t * VN * DD;
      wmma_gemm_kernel<<<gg(EN, DD), blk, 0, stream>>>(
          LoadEdge{Ht, srcp, dstp}, ew[s][0], eb[s][0], StoreDense{bufA, DD}, EN, 256, DD, 1);
      wmma_gemm_kernel<<<gg(EN, DD), blk, 0, stream>>>(
          LoadDense{bufA, DD}, ew[s][1], eb[s][1], StoreDense{bufB, DD}, EN, DD, DD, 1);
      wmma_gemm_kernel<<<gg(EN, DD), blk, 0, stream>>>(
          LoadDense{bufB, DD}, ew[s][2], eb[s][2], StoreDense{bufA, DD}, EN, DD, DD, 0);
      scatter_add_kernel<<<(EN * DD + 255) / 256, 256, 0, stream>>>(bufA, dst_raw, agg, t);
    }
    // node MLP: [h_frames | agg] -> 128 relu -> 128, scatter into (V,T,S,D)
    wmma_gemm_kernel<<<gg(TT * VN, DD), blk, 0, stream>>>(
        LoadConcat{h_frames, agg}, nw[s][0], nb[s][0], StoreDense{bufA, DD}, TT * VN, 256, DD, 1);
    wmma_gemm_kernel<<<gg(TT * VN, DD), blk, 0, stream>>>(
        LoadDense{bufA, DD}, nw[s][1], nb[s][1], StoreMsg{all_msg, s}, TT * VN, DD, DD, 0);
  }

  // spatiotemporal attention (chunked over t; fused concat loader, K=273 zero-padded)
  for (int t = 0; t < TT; ++t) {
    wmma_gemm_kernel<<<gg(VN * SS, DD), blk, 0, stream>>>(
        LoadAttn{h_t, all_msg, theta, t}, attn_w0, attn_b0, StoreDense{bufA, DD},
        VN * SS, 273, DD, 1);
    attn_logits_kernel<<<(VN * SS + 255) / 256, 256, 0, stream>>>(
        bufA, attn_w1, attn_b1, logmask, logits, t);
  }
  softmax_agg_kernel<<<VN, 128, 0, stream>>>(logits, all_msg, agg_msg);

  // dyn MLP: [h_t | agg_msg] 256 -> 256 -> 256 -> 256 -> 128 (tanh) -> 3 (linear + mask)
  wmma_gemm_kernel<<<gg(VN, 256), blk, 0, stream>>>(
      LoadConcat{h_t, agg_msg}, dyn_w[0], dyn_b[0], StoreDense{bufA, 256}, VN, 256, 256, 2);
  wmma_gemm_kernel<<<gg(VN, 256), blk, 0, stream>>>(
      LoadDense{bufA, 256}, dyn_w[1], dyn_b[1], StoreDense{bufB, 256}, VN, 256, 256, 2);
  wmma_gemm_kernel<<<gg(VN, 256), blk, 0, stream>>>(
      LoadDense{bufB, 256}, dyn_w[2], dyn_b[2], StoreDense{bufA, 256}, VN, 256, 256, 2);
  wmma_gemm_kernel<<<gg(VN, DD), blk, 0, stream>>>(
      LoadDense{bufA, 256}, dyn_w[3], dyn_b[3], StoreDense{bufB, DD}, VN, 256, DD, 2);
  final_kernel<<<(VN + 127) / 128, 128, 0, stream>>>(bufB, dyn_w[4], dyn_b[4],
                                                     constraint, (float*)d_out);
}
